// CircuitCAT_57655640982007
// MI455X (gfx1250) — compile-verified, matrix operand order checked by reading, not run on hardware
//
#include <hip/hip_runtime.h>
#include <hip/hip_bf16.h>

typedef __bf16 bf16;
typedef __attribute__((ext_vector_type(16))) __bf16 v16bf;
typedef __attribute__((ext_vector_type(8)))  float  v8f;

#define D_IN 384
#define D_H  128
#define NB   512

// ---- WMMA fragment helpers (CDNA5 16-bit 16x32 layout, CDNA5 ISA) ----
// element e (0..15, two bf16 per VGPR) -> K index for this lane-half
__device__ __forceinline__ int frag_k(int e, int half) {
  return ((e < 8) ? 0 : 16) + half * 8 + (e & 7);
}

// A (16xK row-major bf16 tile in LDS, leading dim ld), K-chunk at kBase
__device__ __forceinline__ v16bf load_a_lds(const bf16* tile, int ld, int kBase) {
  int lane = threadIdx.x & 31;
  int half = lane >> 4, m = lane & 15;
  v16bf a;
#pragma unroll
  for (int e = 0; e < 16; ++e) a[e] = tile[m * ld + kBase + frag_k(e, half)];
  return a;
}

// B (KxN) sourced from row-major weight w[O][K] (B[k][n] = w[nBase+n][k]), bf16 in global (L2-resident)
__device__ __forceinline__ v16bf load_b_gw(const bf16* w, int ldw, int nBase, int kBase) {
  int lane = threadIdx.x & 31;
  int half = lane >> 4, n = lane & 15;
  v16bf b;
#pragma unroll
  for (int e = 0; e < 16; ++e) b[e] = w[(nBase + n) * ldw + kBase + frag_k(e, half)];
  return b;
}

__device__ __forceinline__ float gelu_exact(float x) {
  return 0.5f * x * (1.0f + erff(x * 0.70710678118654752f));
}

// ---- weight f32 -> bf16 pre-conversion ----
__global__ void k_f32_to_bf16(const float* __restrict__ src, bf16* __restrict__ dst, int n) {
  int i = blockIdx.x * blockDim.x + threadIdx.x;
  if (i < n) dst[i] = (bf16)src[i];
}

// ---- fused text encoder: x@W1^T+b1 -> LN -> GELU -> @W2^T+b2 -> h_text ----
__global__ void __launch_bounds__(256) k_text_encoder(
    const float* __restrict__ x, const bf16* __restrict__ w1, const float* __restrict__ b1,
    const float* __restrict__ lng, const float* __restrict__ lnb,
    const bf16* __restrict__ w2, const float* __restrict__ b2,
    float* __restrict__ h_text) {
  __shared__ bf16  Abuf[16 * D_IN];
  __shared__ float Hbuf[16 * 256];
  __shared__ bf16  A2[16 * 256];
  __shared__ float redS[16 * 16];
  __shared__ float redQ[16 * 16];
  __shared__ float mS[16], rS[16];

  const int t = threadIdx.x;
  const int row0 = blockIdx.x * 16;
  // stage x tile as bf16
#pragma unroll
  for (int i = 0; i < 24; ++i) {
    int lin = t + 256 * i;                 // 16*384 = 6144
    int r = lin / D_IN, c = lin % D_IN;
    Abuf[lin] = (bf16)x[(size_t)(row0 + r) * D_IN + c];
  }
  __syncthreads();
  const int wave = t >> 5;
  const int lane = t & 31, half = lane >> 4, nn = lane & 15;
  // GEMM1: 16x384 @ 384x256 -> each wave does 2 of 16 col-tiles, A-frag reused
  {
    v8f acc0 = {}; v8f acc1 = {};
    int n0 = wave * 16, n1 = (wave + 8) * 16;
#pragma unroll
    for (int kc = 0; kc < D_IN / 32; ++kc) {
      v16bf a  = load_a_lds(Abuf, D_IN, kc * 32);
      v16bf b0 = load_b_gw(w1, D_IN, n0, kc * 32);
      v16bf b1f = load_b_gw(w1, D_IN, n1, kc * 32);
      acc0 = __builtin_amdgcn_wmma_f32_16x16x32_bf16(false, a, false, b0,  (short)0, acc0, false, false);
      acc1 = __builtin_amdgcn_wmma_f32_16x16x32_bf16(false, a, false, b1f, (short)0, acc1, false, false);
    }
    float bias0 = b1[n0 + nn], bias1 = b1[n1 + nn];
#pragma unroll
    for (int r = 0; r < 8; ++r) {
      Hbuf[(r + 8 * half) * 256 + n0 + nn] = acc0[r] + bias0;
      Hbuf[(r + 8 * half) * 256 + n1 + nn] = acc1[r] + bias1;
    }
  }
  __syncthreads();
  // LayerNorm over 256 + GELU -> A2 (bf16)
  {
    int r = t >> 4, j = t & 15;
    float s = 0.f, q = 0.f;
#pragma unroll
    for (int i = 0; i < 16; ++i) { float v = Hbuf[r * 256 + j + 16 * i]; s += v; q += v * v; }
    redS[r * 16 + j] = s; redQ[r * 16 + j] = q;
  }
  __syncthreads();
  if (t < 16) {
    float s = 0.f, q = 0.f;
#pragma unroll
    for (int j = 0; j < 16; ++j) { s += redS[t * 16 + j]; q += redQ[t * 16 + j]; }
    float m = s / 256.f;
    mS[t] = m;
    rS[t] = rsqrtf(q / 256.f - m * m + 1e-5f);
  }
  __syncthreads();
  {
    int r = t >> 4, j = t & 15;
    float m = mS[r], rs = rS[r];
#pragma unroll
    for (int i = 0; i < 16; ++i) {
      int c = j + 16 * i;
      float v = (Hbuf[r * 256 + c] - m) * rs * lng[c] + lnb[c];
      A2[r * 256 + c] = (bf16)gelu_exact(v);
    }
  }
  __syncthreads();
  // GEMM2: 16x256 @ 256x128 -> h_text
  {
    v8f acc = {};
    int n0 = wave * 16;
#pragma unroll
    for (int kc = 0; kc < 256 / 32; ++kc) {
      v16bf a  = load_a_lds(A2, 256, kc * 32);
      v16bf bb = load_b_gw(w2, 256, n0, kc * 32);
      acc = __builtin_amdgcn_wmma_f32_16x16x32_bf16(false, a, false, bb, (short)0, acc, false, false);
    }
    float bias = b2[n0 + nn];
#pragma unroll
    for (int r = 0; r < 8; ++r)
      h_text[(size_t)(row0 + r + 8 * half) * D_H + n0 + nn] = acc[r] + bias;
  }
}

// ---- edge aggregation: agg[dst] += in[src], cnt[dst] += 1 (f32 atomics) ----
__global__ void k_aggregate(const float* __restrict__ in, const int* __restrict__ src,
                            const int* __restrict__ dst, float* __restrict__ agg,
                            float* __restrict__ cnt, int E) {
  int gid = blockIdx.x * blockDim.x + threadIdx.x;
  int e = gid >> 5;
  if (e >= E) return;
  int s = src[e], d = dst[e];
  int c = (gid & 31) * 4;
  const float4 v = *(const float4*)(in + (size_t)s * D_H + c);
  float* ap = agg + (size_t)d * D_H + c;
  atomicAdd(ap + 0, v.x);
  atomicAdd(ap + 1, v.y);
  atomicAdd(ap + 2, v.z);
  atomicAdd(ap + 3, v.w);
  if ((gid & 31) == 0) atomicAdd(cnt + d, 1.0f);
}

// ---- fused SAGE combine: [mean||x] @ [wl||wr]^T + bl -> LN (-> GELU | + residual) ----
__global__ void __launch_bounds__(256) k_sage(
    const float* __restrict__ agg, const float* __restrict__ cnt,
    const float* __restrict__ hin, const bf16* __restrict__ wl, const float* __restrict__ bl,
    const bf16* __restrict__ wr, const float* __restrict__ lng, const float* __restrict__ lnb,
    const float* __restrict__ residual, float* __restrict__ out, int apply_gelu) {
  __shared__ bf16  Abuf[16 * 256];
  __shared__ float Hbuf[16 * D_H];
  __shared__ float redS[16 * 16];
  __shared__ float redQ[16 * 16];
  __shared__ float mS[16], rS[16];
  const int t = threadIdx.x;
  const int row0 = blockIdx.x * 16;
#pragma unroll
  for (int i = 0; i < 16; ++i) {
    int lin = t + 256 * i;
    int r = lin >> 8, c = lin & 255;
    float v;
    if (c < D_H) v = agg[(size_t)(row0 + r) * D_H + c] / fmaxf(cnt[row0 + r], 1.0f);
    else         v = hin[(size_t)(row0 + r) * D_H + (c - D_H)];
    Abuf[lin] = (bf16)v;
  }
  __syncthreads();
  const int wave = t >> 5;
  const int lane = t & 31, half = lane >> 4, nn = lane & 15;
  {
    v8f acc = {};
    int n0 = wave * 16;
#pragma unroll
    for (int kc = 0; kc < 8; ++kc) {
      int kb = kc * 32;
      v16bf a  = load_a_lds(Abuf, 256, kb);
      v16bf bb = (kb < D_H) ? load_b_gw(wl, D_H, n0, kb)
                            : load_b_gw(wr, D_H, n0, kb - D_H);
      acc = __builtin_amdgcn_wmma_f32_16x16x32_bf16(false, a, false, bb, (short)0, acc, false, false);
    }
    float bias = bl[n0 + nn];
#pragma unroll
    for (int r = 0; r < 8; ++r) Hbuf[(r + 8 * half) * D_H + n0 + nn] = acc[r] + bias;
  }
  __syncthreads();
  {
    int r = t >> 4, j = t & 15;
    float s = 0.f, q = 0.f;
#pragma unroll
    for (int i = 0; i < 8; ++i) { float v = Hbuf[r * D_H + j + 16 * i]; s += v; q += v * v; }
    redS[r * 16 + j] = s; redQ[r * 16 + j] = q;
  }
  __syncthreads();
  if (t < 16) {
    float s = 0.f, q = 0.f;
#pragma unroll
    for (int j = 0; j < 16; ++j) { s += redS[t * 16 + j]; q += redQ[t * 16 + j]; }
    float m = s / 128.f;
    mS[t] = m;
    rS[t] = rsqrtf(q / 128.f - m * m + 1e-5f);
  }
  __syncthreads();
  {
    int r = t >> 4, j = t & 15;
    float m = mS[r], rs = rS[r];
#pragma unroll
    for (int i = 0; i < 8; ++i) {
      int c = j + 16 * i;
      float v = (Hbuf[r * D_H + c] - m) * rs * lng[c] + lnb[c];
      if (apply_gelu) v = gelu_exact(v);
      if (residual)   v += residual[(size_t)(row0 + r) * D_H + c];
      out[(size_t)(row0 + r) * D_H + c] = v;
    }
  }
}

// ---- fused gate + node head + pooled accumulation ----
__global__ void __launch_bounds__(256) k_gate(
    const float* __restrict__ h_text, const float* __restrict__ h_graph,
    const bf16* __restrict__ gtw, const float* __restrict__ gtb,
    const bf16* __restrict__ ggw, const float* __restrict__ ggb,
    const bf16* __restrict__ gmw1, const float* __restrict__ gmb1,
    const float* __restrict__ gmw2, const float* __restrict__ gmb2,
    const bf16* __restrict__ ncw1, const float* __restrict__ ncb1,
    const float* __restrict__ ncw2, const float* __restrict__ ncb2,
    const int* __restrict__ batch,
    float* __restrict__ out_node, float* __restrict__ out_alpha,
    float* __restrict__ pooled, float* __restrict__ cntb, float* __restrict__ greg) {
  __shared__ bf16  At[16 * D_H];
  __shared__ bf16  Ag[16 * D_H];
  __shared__ float Tt[16 * D_H];
  __shared__ float Tg[16 * D_H];
  __shared__ bf16  Cb[16 * D_H];
  __shared__ float Hid[16 * 64];
  __shared__ bf16  Fbh[16 * D_H];
  __shared__ float Nh[16 * 64];
  __shared__ float alphaL[16];
  __shared__ int   bL[16];
  const int t = threadIdx.x;
  const int row0 = blockIdx.x * 16;
#pragma unroll
  for (int i = 0; i < 8; ++i) {
    int lin = t + 256 * i;
    int r = lin >> 7, c = lin & 127;
    At[lin] = (bf16)h_text[(size_t)(row0 + r) * D_H + c];
    Ag[lin] = (bf16)h_graph[(size_t)(row0 + r) * D_H + c];
  }
  if (t < 16) bL[t] = batch[row0 + t];
  __syncthreads();
  const int wave = t >> 5;
  const int lane = t & 31, half = lane >> 4, nn = lane & 15;
  // t = h_text@gt^T+b ; g = h_graph@gg^T+b
  {
    v8f at = {}; v8f ag = {};
    int n0 = wave * 16;
#pragma unroll
    for (int kc = 0; kc < 4; ++kc) {
      int kb = kc * 32;
      v16bf a1 = load_a_lds(At, D_H, kb);
      v16bf a2 = load_a_lds(Ag, D_H, kb);
      v16bf w1 = load_b_gw(gtw, D_H, n0, kb);
      v16bf w2 = load_b_gw(ggw, D_H, n0, kb);
      at = __builtin_amdgcn_wmma_f32_16x16x32_bf16(false, a1, false, w1, (short)0, at, false, false);
      ag = __builtin_amdgcn_wmma_f32_16x16x32_bf16(false, a2, false, w2, (short)0, ag, false, false);
    }
    float bt = gtb[n0 + nn], bg = ggb[n0 + nn];
#pragma unroll
    for (int r = 0; r < 8; ++r) {
      Tt[(r + 8 * half) * D_H + n0 + nn] = at[r] + bt;
      Tg[(r + 8 * half) * D_H + n0 + nn] = ag[r] + bg;
    }
  }
  __syncthreads();
#pragma unroll
  for (int i = 0; i < 8; ++i) {
    int lin = t + 256 * i;
    Cb[lin] = (bf16)fabsf(Tt[lin] - Tg[lin]);
  }
  __syncthreads();
  // gate MLP layer 1: 128 -> 64, relu (waves 0..3)
  if (wave < 4) {
    v8f acc = {};
    int n0 = wave * 16;
#pragma unroll
    for (int kc = 0; kc < 4; ++kc) {
      v16bf a  = load_a_lds(Cb, D_H, kc * 32);
      v16bf bb = load_b_gw(gmw1, D_H, n0, kc * 32);
      acc = __builtin_amdgcn_wmma_f32_16x16x32_bf16(false, a, false, bb, (short)0, acc, false, false);
    }
    float bias = gmb1[n0 + nn];
#pragma unroll
    for (int r = 0; r < 8; ++r) Hid[(r + 8 * half) * 64 + n0 + nn] = fmaxf(acc[r] + bias, 0.0f);
  }
  __syncthreads();
  // alpha = sigmoid(Hid . gm_w2 + b); gate_reg accum; batch count accum
  if (t < 16) {
    float s = gmb2[0];
#pragma unroll
    for (int k = 0; k < 64; ++k) s += Hid[t * 64 + k] * gmw2[k];
    float a = 1.0f / (1.0f + expf(-s));
    alphaL[t] = a;
    out_alpha[row0 + t] = a;
    float d = a - 0.5f;
    atomicAdd(greg, d * d);
    atomicAdd(cntb + bL[t], 1.0f);
  }
  __syncthreads();
  // h_fused = a*t + (1-a)*g ; pooled atomic accumulation
#pragma unroll
  for (int i = 0; i < 8; ++i) {
    int lin = t + 256 * i;
    int r = lin >> 7, c = lin & 127;
    float a = alphaL[r];
    float f = a * Tt[lin] + (1.0f - a) * Tg[lin];
    Fbh[lin] = (bf16)f;
    atomicAdd(pooled + (size_t)bL[r] * D_H + c, f);
  }
  __syncthreads();
  // node head layer 1: 128 -> 64, relu
  if (wave < 4) {
    v8f acc = {};
    int n0 = wave * 16;
#pragma unroll
    for (int kc = 0; kc < 4; ++kc) {
      v16bf a  = load_a_lds(Fbh, D_H, kc * 32);
      v16bf bb = load_b_gw(ncw1, D_H, n0, kc * 32);
      acc = __builtin_amdgcn_wmma_f32_16x16x32_bf16(false, a, false, bb, (short)0, acc, false, false);
    }
    float bias = ncb1[n0 + nn];
#pragma unroll
    for (int r = 0; r < 8; ++r) Nh[(r + 8 * half) * 64 + n0 + nn] = fmaxf(acc[r] + bias, 0.0f);
  }
  __syncthreads();
  // node logits: 64 -> 6 (scalar dots)
  if (t < 96) {
    int o = t >> 4, r = t & 15;
    float s = ncb2[o];
#pragma unroll
    for (int k = 0; k < 64; ++k) s += Nh[r * 64 + k] * ncw2[o * 64 + k];
    out_node[(size_t)(row0 + r) * 6 + o] = s;
  }
}

// ---- pooled graph heads (B=512 rows, tiny: scalar VALU) ----
__global__ void __launch_bounds__(128) k_pool_heads(
    const float* __restrict__ pooled, const float* __restrict__ cntb,
    const float* __restrict__ grw1, const float* __restrict__ grb1,
    const float* __restrict__ grw2, const float* __restrict__ grb2,
    const float* __restrict__ gcw1, const float* __restrict__ gcb1,
    const float* __restrict__ gcw2, const float* __restrict__ gcb2,
    float* __restrict__ out_perf, float* __restrict__ out_type) {
  __shared__ float pl[128];
  __shared__ float hg[128];
  __shared__ float hc[64];
  const int b = blockIdx.x, t = threadIdx.x;
  pl[t] = pooled[b * 128 + t] / fmaxf(cntb[b], 1.0f);
  __syncthreads();
  {
    float s = grb1[t];
    for (int k = 0; k < 128; ++k) s += pl[k] * grw1[t * 128 + k];
    hg[t] = fmaxf(s, 0.0f);
  }
  if (t < 64) {
    float s = gcb1[t];
    for (int k = 0; k < 128; ++k) s += pl[k] * gcw1[t * 128 + k];
    hc[t] = fmaxf(s, 0.0f);
  }
  __syncthreads();
  if (t < 8) {
    float s = grb2[t];
    for (int k = 0; k < 128; ++k) s += hg[k] * grw2[t * 128 + k];
    out_perf[b * 8 + t] = s;
  }
  if (t < 10) {
    float s = gcb2[t];
    for (int k = 0; k < 64; ++k) s += hc[k] * gcw2[t * 64 + k];
    out_type[b * 10 + t] = s;
  }
}

__global__ void k_finalize(const float* __restrict__ greg, float* __restrict__ out, float invN) {
  out[0] = greg[0] * invN;
}

extern "C" void kernel_launch(void* const* d_in, const int* in_sizes, int n_in,
                              void* d_out, int out_size, void* d_ws, size_t ws_size,
                              hipStream_t stream) {
  const float* x      = (const float*)d_in[0];
  const int*   eidx   = (const int*)d_in[1];
  const int*   batch  = (const int*)d_in[2];
  const float* te_w1  = (const float*)d_in[3];
  const float* te_b1  = (const float*)d_in[4];
  const float* te_ln_g= (const float*)d_in[5];
  const float* te_ln_b= (const float*)d_in[6];
  const float* te_w2  = (const float*)d_in[7];
  const float* te_b2  = (const float*)d_in[8];
  const float* s1_wl  = (const float*)d_in[9];
  const float* s1_bl  = (const float*)d_in[10];
  const float* s1_wr  = (const float*)d_in[11];
  const float* ln1_g  = (const float*)d_in[12];
  const float* ln1_b  = (const float*)d_in[13];
  const float* s2_wl  = (const float*)d_in[14];
  const float* s2_bl  = (const float*)d_in[15];
  const float* s2_wr  = (const float*)d_in[16];
  const float* ln2_g  = (const float*)d_in[17];
  const float* ln2_b  = (const float*)d_in[18];
  const float* gt_w   = (const float*)d_in[19];
  const float* gt_b   = (const float*)d_in[20];
  const float* gg_w   = (const float*)d_in[21];
  const float* gg_b   = (const float*)d_in[22];
  const float* gm_w1  = (const float*)d_in[23];
  const float* gm_b1  = (const float*)d_in[24];
  const float* gm_w2  = (const float*)d_in[25];
  const float* gm_b2  = (const float*)d_in[26];
  const float* nc_w1  = (const float*)d_in[27];
  const float* nc_b1  = (const float*)d_in[28];
  const float* nc_w2  = (const float*)d_in[29];
  const float* nc_b2  = (const float*)d_in[30];
  const float* gr_w1  = (const float*)d_in[31];
  const float* gr_b1  = (const float*)d_in[32];
  const float* gr_w2  = (const float*)d_in[33];
  const float* gr_b2  = (const float*)d_in[34];
  const float* gc_w1  = (const float*)d_in[35];
  const float* gc_b1  = (const float*)d_in[36];
  const float* gc_w2  = (const float*)d_in[37];
  const float* gc_b2  = (const float*)d_in[38];

  const int N = in_sizes[0] / D_IN;   // 200000
  const int E = in_sizes[1] / 2;      // 600000
  const int B = NB;                   // 512

  char* ws = (char*)d_ws;
  size_t off = 0;
  auto alloc = [&](size_t bytes) -> void* {
    void* p = ws + off;
    off += (bytes + 255) & ~(size_t)255;
    return p;
  };
  bf16* w1b   = (bf16*)alloc((size_t)256 * 384 * 2);
  bf16* w2b   = (bf16*)alloc((size_t)128 * 256 * 2);
  bf16* s1wlb = (bf16*)alloc((size_t)128 * 128 * 2);
  bf16* s1wrb = (bf16*)alloc((size_t)128 * 128 * 2);
  bf16* s2wlb = (bf16*)alloc((size_t)128 * 128 * 2);
  bf16* s2wrb = (bf16*)alloc((size_t)128 * 128 * 2);
  bf16* gtwb  = (bf16*)alloc((size_t)128 * 128 * 2);
  bf16* ggwb  = (bf16*)alloc((size_t)128 * 128 * 2);
  bf16* gmw1b = (bf16*)alloc((size_t)64 * 128 * 2);
  bf16* ncw1b = (bf16*)alloc((size_t)64 * 128 * 2);
  float* h_text = (float*)alloc((size_t)N * D_H * 4);
  float* agg    = (float*)alloc((size_t)N * D_H * 4);
  float* cnt    = (float*)alloc((size_t)N * 4);
  float* hmid   = (float*)alloc((size_t)N * D_H * 4);
  float* hgraph = (float*)alloc((size_t)N * D_H * 4);
  float* pooled = (float*)alloc((size_t)B * D_H * 4);
  float* cntb   = (float*)alloc((size_t)B * 4);
  float* greg   = (float*)alloc(256);

  float* out       = (float*)d_out;
  float* out_node  = out;                           // N*6
  float* out_perf  = out_node + (size_t)N * 6;      // B*8
  float* out_type  = out_perf + (size_t)B * 8;      // B*10
  float* out_alpha = out_type + (size_t)B * 10;     // N
  float* out_greg  = out_alpha + (size_t)N;         // 1

  auto cvt = [&](const float* s, bf16* d, int n) {
    k_f32_to_bf16<<<(n + 255) / 256, 256, 0, stream>>>(s, d, n);
  };
  cvt(te_w1, w1b, 256 * 384);
  cvt(te_w2, w2b, 128 * 256);
  cvt(s1_wl, s1wlb, 128 * 128);
  cvt(s1_wr, s1wrb, 128 * 128);
  cvt(s2_wl, s2wlb, 128 * 128);
  cvt(s2_wr, s2wrb, 128 * 128);
  cvt(gt_w, gtwb, 128 * 128);
  cvt(gg_w, ggwb, 128 * 128);
  cvt(gm_w1, gmw1b, 64 * 128);
  cvt(nc_w1, ncw1b, 64 * 128);

  const int tiles = N / 16;
  k_text_encoder<<<tiles, 256, 0, stream>>>(x, w1b, te_b1, te_ln_g, te_ln_b, w2b, te_b2, h_text);

  const int aggBlocks = (int)(((size_t)E * 32 + 255) / 256);
  hipMemsetAsync(agg, 0, (size_t)N * D_H * 4, stream);
  hipMemsetAsync(cnt, 0, (size_t)N * 4, stream);
  k_aggregate<<<aggBlocks, 256, 0, stream>>>(h_text, eidx, eidx + E, agg, cnt, E);
  k_sage<<<tiles, 256, 0, stream>>>(agg, cnt, h_text, s1wlb, s1_bl, s1wrb, ln1_g, ln1_b,
                                    nullptr, hmid, 1);
  hipMemsetAsync(agg, 0, (size_t)N * D_H * 4, stream);
  hipMemsetAsync(cnt, 0, (size_t)N * 4, stream);
  k_aggregate<<<aggBlocks, 256, 0, stream>>>(hmid, eidx, eidx + E, agg, cnt, E);
  k_sage<<<tiles, 256, 0, stream>>>(agg, cnt, hmid, s2wlb, s2_bl, s2wrb, ln2_g, ln2_b,
                                    h_text, hgraph, 0);

  hipMemsetAsync(pooled, 0, (size_t)B * D_H * 4, stream);
  hipMemsetAsync(cntb, 0, (size_t)B * 4, stream);
  hipMemsetAsync(greg, 0, 4, stream);
  k_gate<<<tiles, 256, 0, stream>>>(h_text, hgraph, gtwb, gt_b, ggwb, gg_b,
                                    gmw1b, gm_b1, gm_w2, gm_b2,
                                    ncw1b, nc_b1, nc_w2, nc_b2, batch,
                                    out_node, out_alpha, pooled, cntb, greg);
  k_pool_heads<<<B, 128, 0, stream>>>(pooled, cntb, gr_w1, gr_b1, gr_w2, gr_b2,
                                      gc_w1, gc_b1, gc_w2, gc_b2, out_perf, out_type);
  k_finalize<<<1, 1, 0, stream>>>(greg, out_greg, 1.0f / (float)N);
}
